// MoAETransformer_68745246540189
// MI455X (gfx1250) — matrix operand order repaired and checked by reading, not compiled, not run on hardware
//
#include <hip/hip_runtime.h>

typedef __bf16 bf16;
typedef __attribute__((ext_vector_type(16))) __bf16 v16bf;
typedef __attribute__((ext_vector_type(8)))  float  v8f;
typedef __attribute__((ext_vector_type(4)))  unsigned int v4u;
typedef __attribute__((ext_vector_type(8)))  int v8i;
typedef __attribute__((ext_vector_type(4)))  int v4i;

#if defined(__has_builtin)
#if __has_builtin(__builtin_amdgcn_tensor_load_to_lds) && \
    __has_builtin(__builtin_amdgcn_s_wait_tensorcnt)
#define USE_TDM 1
#endif
#endif

#define DV 768
#define D3 2304
#define VV 32000
#define LV 2
#define EV 8
#define HV 12
#define BV 4
#define SV 1024
#define CV 320
#define NTOK (BV * SV)   // 4096

// ---------------------------------------------------------------------------
// WMMA fragment helpers (bf16 16x16x32, wave32)
// A-matrix 16x32 (MxK): lanes 0-15 row=lane, K 0..15; lanes 16-31 row=lane-16, K 16..31
// B-matrix 32x16 (KxN) from W[N][K] row-major: lanes 0-15 col=lane K 0..15; lanes 16-31 col=lane-16 K 16..31
// C/D 16x16: VGPR i -> row m0+i (m0 = (lane>=16)*8), col = lane&15
// ---------------------------------------------------------------------------
__device__ __forceinline__ int laneid() { return threadIdx.x & 31; }

__device__ __forceinline__ v16bf frag_a(const bf16* p, int ld) {
  int lane = laneid();
  const bf16* q = p + (lane & 15) * ld + (((lane >> 4) & 1) << 4);
  v16bf f;
#pragma unroll
  for (int i = 0; i < 8; ++i) f[i] = q[i];
#pragma unroll
  for (int i = 0; i < 8; ++i) f[8 + i] = q[8 + i];
  return f;
}

__device__ __forceinline__ v16bf frag_b(const bf16* p, int ld) {
  int lane = laneid();
  const bf16* q = p + (lane & 15) * ld + (((lane >> 4) & 1) << 4);
  v16bf f;
#pragma unroll
  for (int i = 0; i < 16; ++i) f[i] = q[i];
  return f;
}

__device__ __forceinline__ v8f wmma_bf16(v16bf a, v16bf b, v8f c) {
  return __builtin_amdgcn_wmma_f32_16x16x32_bf16(false, a, false, b, (short)0, c,
                                                 false, false);
}

// ---------------------------------------------------------------------------
// Generic batched GEMM: C[z] = A[z] @ W[z%wmod]^T + bias[z%wmod] (+ Res[z])
// Block tile 128(M) x 128(N) x 32(K); 8 waves, wave tile 32x64 -> 8 WMMA/K-step
// Weight tile fetched by the Tensor Data Mover (TENSORcnt) when available;
// activations staged+converted by the ALUs in parallel.
// ---------------------------------------------------------------------------
__global__ __launch_bounds__(256) void k_gemm(
    const float* __restrict__ A, const float* __restrict__ W,
    const float* __restrict__ bias, const float* __restrict__ Res,
    float* __restrict__ C, int M, int N, int Kd, long sA, long sW, long sB,
    long sC, int wmod) {
  __shared__ bf16 As[128][40];
  __shared__ bf16 Bs[128][40];
#ifdef USE_TDM
  __shared__ float Bs32[128 * 32];
#endif
  int z = blockIdx.z;
  int wi = z % wmod;
  A += (long)z * sA;
  W += (long)wi * sW;
  const float* bptr = bias + (long)wi * sB;
  C += (long)z * sC;
  if (Res) Res += (long)z * sC;

  int tid = threadIdx.x;
  int wid = tid >> 5;
  int wm = wid & 3, wn = wid >> 2;   // 4 x 32 rows, 2 x 64 cols
  int rowBase = blockIdx.y * 128;
  int colBase = blockIdx.x * 128;

  const v8f z8 = {0.f, 0.f, 0.f, 0.f, 0.f, 0.f, 0.f, 0.f};
  v8f acc0[4], acc1[4];
#pragma unroll
  for (int j = 0; j < 4; ++j) { acc0[j] = z8; acc1[j] = z8; }

  int sr = tid >> 1, sc = (tid & 1) << 4;   // 16 contiguous floats / thread
  int gr = rowBase + sr;
  if (gr > M - 1) gr = M - 1;               // clamp ragged M (experts: M=320)

  int nk = Kd >> 5;
  for (int kb = 0; kb < nk; ++kb) {
    int k0 = kb << 5;
    __syncthreads();
    // stage A (fp32 -> bf16)
    const float* ag = A + (long)gr * Kd + k0 + sc;
#pragma unroll
    for (int i = 0; i < 16; ++i) As[sr][sc + i] = (bf16)ag[i];
#ifdef USE_TDM
    // Tensor Data Mover: DMA the 128x32 fp32 weight tile into LDS (D# per
    // CDNA5 ISA: count=1, type=2, data_size=4B,
    // tile_dim0=32, tile_dim1=128, tensor_dim0_stride=Kd).
    if (wid == 0) {
      unsigned lds_off = (unsigned)(size_t)(void*)Bs32;
      unsigned long long ga =
          (unsigned long long)(const void*)(W + (long)colBase * Kd + k0);
      v4u g0 = {1u, lds_off, (unsigned)ga,
                (unsigned)((ga >> 32) & 0x1ffffffu) | (2u << 30)};
      v8i g1 = {(int)(2u << 16),                                  // data_size=4B
                (int)(((unsigned)Kd & 0xffffu) << 16),            // tensor_dim0 lo
                (int)(((unsigned)Kd >> 16) | (((unsigned)N & 0xffffu) << 16)),
                (int)(((unsigned)N >> 16) | (32u << 16)),         // tile_dim0=32
                (int)128,                                         // tile_dim1=128
                (int)(unsigned)Kd,                                // dim0 stride lo
                0, 0};
      v4i gz = {0, 0, 0, 0};
      v8i gz8 = {0, 0, 0, 0, 0, 0, 0, 0};
      __builtin_amdgcn_tensor_load_to_lds(g0, g1, gz, gz, gz8, 0);
      __builtin_amdgcn_s_wait_tensorcnt(0);
    }
    __syncthreads();
    // convert DMA'd fp32 tile -> bf16 WMMA layout
    {
      const float* src = Bs32 + sr * 32 + sc;
#pragma unroll
      for (int i = 0; i < 16; ++i) Bs[sr][sc + i] = (bf16)src[i];
    }
    if (kb + 1 < nk) __builtin_prefetch(ag + 32, 0, 1);
#else
    const float* wg = W + (long)(colBase + sr) * Kd + k0 + sc;
#pragma unroll
    for (int i = 0; i < 16; ++i) Bs[sr][sc + i] = (bf16)wg[i];
    if (kb + 1 < nk) {  // global_prefetch_b8 next K tile
      __builtin_prefetch(ag + 32, 0, 1);
      __builtin_prefetch(wg + 32, 0, 1);
    }
#endif
    __syncthreads();
    v16bf a0 = frag_a(&As[wm * 32][0], 40);
    v16bf a1 = frag_a(&As[wm * 32 + 16][0], 40);
#pragma unroll
    for (int j = 0; j < 4; ++j) {
      v16bf b = frag_b(&Bs[wn * 64 + j * 16][0], 40);
      acc0[j] = wmma_bf16(a0, b, acc0[j]);
      acc1[j] = wmma_bf16(a1, b, acc1[j]);
    }
  }

  int lane = laneid();
  int n = lane & 15;
  int m0 = ((lane >> 4) & 1) << 3;
#pragma unroll
  for (int j = 0; j < 4; ++j) {
    int gcol = colBase + wn * 64 + j * 16 + n;
    float bv = bptr[gcol];
#pragma unroll
    for (int mi = 0; mi < 2; ++mi) {
      v8f* ac = mi ? acc1 : acc0;
#pragma unroll
      for (int i = 0; i < 8; ++i) {
        long r = rowBase + wm * 32 + mi * 16 + m0 + i;
        if (r < M) {
          float v = ac[j][i] + bv;
          if (Res) v += Res[r * N + gcol];
          C[r * N + gcol] = v;
        }
      }
    }
  }
}

// ---------------------------------------------------------------------------
// Flash attention, one wave per (batch z, head h, 16-query tile).
// qkv rows of length 3*DV: q | k | v, head slice of 64. dh=64, scale=0.125.
// ---------------------------------------------------------------------------
__global__ __launch_bounds__(32) void k_attn(const float* __restrict__ QKV,
                                             float* __restrict__ Out, int seq,
                                             long sQKV, long sOut, int causal) {
  __shared__ bf16 qs[16][72];
  __shared__ bf16 ks[32][72];
  __shared__ bf16 vt[64][36];
  __shared__ bf16 ps[16][40];
  const float scale = 0.125f;  // 1/sqrt(64)
  int h = blockIdx.y;
  int z = blockIdx.z;
  const float* qkv = QKV + (long)z * sQKV;
  float* out = Out + (long)z * sOut;
  int qbase = blockIdx.x * 16;
  int lane = threadIdx.x;
  int qoff = h * 64, koff = DV + h * 64, voff = 2 * DV + h * 64;

  {  // stage Q tile 16x64 -> LDS bf16
    int r = lane >> 1, c0 = (lane & 1) << 5;
    const float* src = qkv + (long)(qbase + r) * D3 + qoff + c0;
#pragma unroll
    for (int c = 0; c < 32; ++c) qs[r][c0 + c] = (bf16)src[c];
  }
  __syncthreads();
  v16bf aq0 = frag_a(&qs[0][0], 72);
  v16bf aq1 = frag_a(&qs[0][32], 72);

  const v8f z8 = {0.f, 0.f, 0.f, 0.f, 0.f, 0.f, 0.f, 0.f};
  v8f o0 = z8, o1 = z8, o2 = z8, o3 = z8;
  float mrun[8], lrun[8];
#pragma unroll
  for (int i = 0; i < 8; ++i) { mrun[i] = -1e30f; lrun[i] = 0.f; }

  int m0 = ((lane >> 4) & 1) << 3;
  int ln = lane & 15;
  int nblk = causal ? ((qbase + 15) >> 5) + 1 : (seq >> 5);
  for (int kb = 0; kb < nblk; ++kb) {
    int kbase = kb << 5;
    __syncthreads();
    {  // stage K (row-major) and V (transposed) blocks
      const float* ksrc = qkv + (long)(kbase + lane) * D3 + koff;
      const float* vsrc = qkv + (long)(kbase + lane) * D3 + voff;
#pragma unroll
      for (int c = 0; c < 64; ++c) {
        ks[lane][c] = (bf16)ksrc[c];
        vt[c][lane] = (bf16)vsrc[c];
      }
    }
    __syncthreads();
    v8f s0 = z8, s1 = z8;
    s0 = wmma_bf16(aq0, frag_b(&ks[0][0], 72), s0);
    s0 = wmma_bf16(aq1, frag_b(&ks[0][32], 72), s0);
    s1 = wmma_bf16(aq0, frag_b(&ks[16][0], 72), s1);
    s1 = wmma_bf16(aq1, frag_b(&ks[16][32], 72), s1);

    float sf[8];
#pragma unroll
    for (int i = 0; i < 8; ++i) {
      float v0 = s0[i] * scale;
      float v1 = s1[i] * scale;
      if (causal) {
        int qrow = qbase + m0 + i;
        if (kbase + ln > qrow) v0 = -1e30f;
        if (kbase + 16 + ln > qrow) v1 = -1e30f;
      }
      float mx = fmaxf(v0, v1);
#pragma unroll
      for (int off = 8; off >= 1; off >>= 1)
        mx = fmaxf(mx, __shfl_xor(mx, off, 32));
      float mnew = fmaxf(mrun[i], mx);
      float f = __expf(mrun[i] - mnew);
      mrun[i] = mnew;
      v0 = __expf(v0 - mnew);
      v1 = __expf(v1 - mnew);
      float rs = v0 + v1;
#pragma unroll
      for (int off = 8; off >= 1; off >>= 1) rs += __shfl_xor(rs, off, 32);
      lrun[i] = lrun[i] * f + rs;
      sf[i] = f;
      ps[m0 + i][ln] = (bf16)v0;          // P in A-fragment source layout
      ps[m0 + i][16 + ln] = (bf16)v1;
    }
#pragma unroll
    for (int i = 0; i < 8; ++i) {
      o0[i] *= sf[i]; o1[i] *= sf[i]; o2[i] *= sf[i]; o3[i] *= sf[i];
    }
    __syncthreads();
    v16bf ap = frag_a(&ps[0][0], 40);
    o0 = wmma_bf16(ap, frag_b(&vt[0][0], 36), o0);
    o1 = wmma_bf16(ap, frag_b(&vt[16][0], 36), o1);
    o2 = wmma_bf16(ap, frag_b(&vt[32][0], 36), o2);
    o3 = wmma_bf16(ap, frag_b(&vt[48][0], 36), o3);
  }
#pragma unroll
  for (int i = 0; i < 8; ++i) {
    float inv = 1.f / lrun[i];
    long r = qbase + m0 + i;
    float* dst = out + r * DV + h * 64;
    dst[ln] = o0[i] * inv;
    dst[16 + ln] = o1[i] * inv;
    dst[32 + ln] = o2[i] * inv;
    dst[48 + ln] = o3[i] * inv;
  }
}

// ---------------------------------------------------------------------------
// Scalar / elementwise kernels
// ---------------------------------------------------------------------------
__global__ __launch_bounds__(256) void k_embed(const int* __restrict__ tokens,
                                               const float* __restrict__ emb,
                                               const float* __restrict__ pos,
                                               float* __restrict__ x) {
  int idx = blockIdx.x;
  int s = idx & (SV - 1);
  int tok = tokens[idx];
  const float* es = emb + (long)tok * DV;
  const float* psr = pos + (long)s * DV;
  float* dst = x + (long)idx * DV;
  for (int d = threadIdx.x; d < DV; d += 256) dst[d] = es[d] + psr[d];
}

__global__ __launch_bounds__(256) void k_ln(const float* __restrict__ in,
                                            const float* __restrict__ g,
                                            const float* __restrict__ bta,
                                            float* __restrict__ outp) {
  int wid = threadIdx.x >> 5;
  int lane = threadIdx.x & 31;
  long row = (long)blockIdx.x * 8 + wid;
  const float* src = in + row * DV;
  float vals[24];
  float sum = 0.f, sq = 0.f;
#pragma unroll
  for (int i = 0; i < 24; ++i) {
    float v = src[lane + i * 32];
    vals[i] = v; sum += v; sq += v * v;
  }
#pragma unroll
  for (int off = 16; off >= 1; off >>= 1) {
    sum += __shfl_xor(sum, off, 32);
    sq += __shfl_xor(sq, off, 32);
  }
  float mean = sum * (1.f / DV);
  float var = sq * (1.f / DV) - mean * mean;
  float inv = rsqrtf(var + 1e-5f);
  float* dst = outp + row * DV;
#pragma unroll
  for (int i = 0; i < 24; ++i) {
    int d = lane + i * 32;
    dst[d] = (vals[i] - mean) * inv * g[d] + bta[d];
  }
}

__global__ __launch_bounds__(256) void k_router(
    const float* __restrict__ x, const float* __restrict__ Wr, int* e1, int* e2,
    float* g1, float* g2, float* acc) {
  int wid = threadIdx.x >> 5;
  int lane = threadIdx.x & 31;
  int tok = blockIdx.x * 8 + wid;
  const float* xv = x + (long)tok * DV;
  float s[8];
#pragma unroll
  for (int e = 0; e < 8; ++e) s[e] = 0.f;
  for (int d = lane; d < DV; d += 32) {
    float xd = xv[d];
#pragma unroll
    for (int e = 0; e < 8; ++e) s[e] += xd * Wr[d * 8 + e];
  }
#pragma unroll
  for (int e = 0; e < 8; ++e)
#pragma unroll
    for (int off = 16; off >= 1; off >>= 1) s[e] += __shfl_xor(s[e], off, 32);
  if (lane == 0) {
    float mx = s[0];
    for (int e = 1; e < 8; ++e) mx = fmaxf(mx, s[e]);
    float p[8], sum = 0.f;
    for (int e = 0; e < 8; ++e) { p[e] = __expf(s[e] - mx); sum += p[e]; }
    float inv = 1.f / sum;
    for (int e = 0; e < 8; ++e) p[e] *= inv;
    float lse = mx + __logf(sum);
    int t1 = 0;
    for (int e = 1; e < 8; ++e) if (p[e] > p[t1]) t1 = e;
    int t2 = (t1 == 0) ? 1 : 0;
    for (int e = 0; e < 8; ++e) if (e != t1 && p[e] > p[t2]) t2 = e;
    e1[tok] = t1; e2[tok] = t2; g1[tok] = p[t1]; g2[tok] = p[t2];
    for (int e = 0; e < 8; ++e) atomicAdd(&acc[e], p[e]);
    atomicAdd(&acc[8 + t1], 1.f);
    atomicAdd(&acc[16], lse * lse);
  }
}

// slot-major capacity scan (top-1 assignments take priority), one thread/batch
__global__ void k_scan(const int* e1, const int* e2, int* p1, int* p2) {
  int b = threadIdx.x;
  if (b >= BV) return;
  int cnt[8];
  for (int e = 0; e < 8; ++e) cnt[e] = 0;
  for (int k = 0; k < 2; ++k) {
    const int* ei = k ? e2 : e1;
    int* pi = k ? p2 : p1;
    for (int n = 0; n < SV; ++n) {
      int idx = b * SV + n;
      int c = cnt[ei[idx]]++;
      pi[idx] = (c < CV) ? c : -1;
    }
  }
}

__global__ __launch_bounds__(256) void k_gather(const float* __restrict__ x,
                                                const int* e1, const int* e2,
                                                const int* p1, const int* p2,
                                                float* __restrict__ xe) {
  int j = blockIdx.x;  // B*S*2 assignments
  int k = j & 1;
  int idx = j >> 1;
  int b = idx >> 10;
  int pos = k ? p2[idx] : p1[idx];
  if (pos < 0) return;
  int e = k ? e2[idx] : e1[idx];
  const float* src = x + (long)idx * DV;
  float* dst = xe + ((long)(b * EV + e) * CV + pos) * DV;
  for (int d = threadIdx.x; d < DV; d += 256) dst[d] = src[d];
}

__global__ __launch_bounds__(256) void k_combine(
    const float* __restrict__ x, const float* __restrict__ ey, const int* e1,
    const int* e2, const int* p1, const int* p2, const float* g1,
    const float* g2, float* __restrict__ t) {
  int idx = blockIdx.x;
  int b = idx >> 10;
  int pa = p1[idx], pb = p2[idx];
  float ga = g1[idx], gb = g2[idx];
  const float* ya =
      (pa >= 0) ? ey + ((long)(b * EV + e1[idx]) * CV + pa) * DV : nullptr;
  const float* yb =
      (pb >= 0) ? ey + ((long)(b * EV + e2[idx]) * CV + pb) * DV : nullptr;
  for (int d = threadIdx.x; d < DV; d += 256) {
    float v = x[(long)idx * DV + d];
    if (ya) v += ga * ya[d];
    if (yb) v += gb * yb[d];
    t[(long)idx * DV + d] = v;
  }
}

__global__ void k_zero(float* p, long n) {
  long i = (long)blockIdx.x * blockDim.x + threadIdx.x;
  long st = (long)gridDim.x * blockDim.x;
  for (; i < n; i += st) p[i] = 0.f;
}

__global__ void k_aux(const float* acc, float* auxp) {
  float bal = 0.f;
  for (int e = 0; e < 8; ++e)
    bal += (acc[e] / (float)NTOK) * (acc[8 + e] / (float)NTOK);
  bal *= 0.01f * 8.f;
  float zl = 0.001f * acc[16] / (float)NTOK;
  *auxp += bal + zl;
}

__global__ void k_auxstore(const float* auxp, float* outp) { outp[0] = auxp[0]; }

// ---------------------------------------------------------------------------
extern "C" void kernel_launch(void* const* d_in, const int* in_sizes, int n_in,
                              void* d_out, int out_size, void* d_ws,
                              size_t ws_size, hipStream_t stream) {
  (void)in_sizes; (void)n_in; (void)out_size; (void)ws_size;
  const int* tokens = (const int*)d_in[0];
  const float* emb = (const float*)d_in[1];
  const float* pos = (const float*)d_in[2];
  const float* aWqkv = (const float*)d_in[3];
  const float* abqkv = (const float*)d_in[4];
  const float* aWo = (const float*)d_in[5];
  const float* abo = (const float*)d_in[6];
  const float* n1g = (const float*)d_in[7];
  const float* n1b = (const float*)d_in[8];
  const float* n2g = (const float*)d_in[9];
  const float* n2b = (const float*)d_in[10];
  const float* routerW = (const float*)d_in[11];
  const float* eWqkv = (const float*)d_in[12];
  const float* ebqkv = (const float*)d_in[13];
  const float* eWo = (const float*)d_in[14];
  const float* ebo = (const float*)d_in[15];
  const float* headW = (const float*)d_in[16];
  const float* headb = (const float*)d_in[17];
  float* out = (float*)d_out;
  float* ws = (float*)d_ws;

  // workspace layout (floats); expert qkv / attn-out buffers alias MHA ones
  const long OX = 0;
  const long OT = OX + (long)NTOK * DV;
  const long OQKV = OT + (long)NTOK * DV;
  const long OATT = OQKV + (long)BV * EV * CV * D3;
  const long OXE = OATT + (long)BV * EV * CV * DV;
  const long OEY = OXE + (long)BV * EV * CV * DV;
  const long OMISC = OEY + (long)BV * EV * CV * DV;

  float* x = ws + OX;
  float* t = ws + OT;
  float* qkv = ws + OQKV;
  float* att = ws + OATT;
  float* xe = ws + OXE;
  float* ey = ws + OEY;
  int* e1 = (int*)(ws + OMISC);
  int* e2 = e1 + NTOK;
  int* p1 = e2 + NTOK;
  int* p2 = p1 + NTOK;
  float* g1 = (float*)(p2 + NTOK);
  float* g2 = g1 + NTOK;
  float* acc = g2 + NTOK;   // me[8], ce[8], z[1]
  float* auxp = acc + 24;

  k_zero<<<1, 256, 0, stream>>>(acc, 32);
  k_embed<<<dim3(NTOK), dim3(256), 0, stream>>>(tokens, emb, pos, x);

  for (int l = 0; l < LV; ++l) {
    // MHA: qkv = x @ aWqkv^T + b
    k_gemm<<<dim3(D3 / 128, NTOK / 128, 1), dim3(256), 0, stream>>>(
        x, aWqkv + (long)l * D3 * DV, abqkv + (long)l * D3, nullptr, qkv, NTOK,
        D3, DV, 0, 0, 0, 0, 1);
    k_attn<<<dim3(SV / 16, HV, BV), dim3(32), 0, stream>>>(
        qkv, att, SV, (long)SV * D3, (long)SV * DV, 1);
    // t = att @ aWo^T + abo + x ; x = LN1(t)
    k_gemm<<<dim3(DV / 128, NTOK / 128, 1), dim3(256), 0, stream>>>(
        att, aWo + (long)l * DV * DV, abo + (long)l * DV, x, t, NTOK, DV, DV, 0,
        0, 0, 0, 1);
    k_ln<<<dim3(NTOK / 8), dim3(256), 0, stream>>>(t, n1g + (long)l * DV,
                                                   n1b + (long)l * DV, x);
    // MoAE: router -> scan -> gather
    k_zero<<<1, 256, 0, stream>>>(acc, 17);
    k_router<<<dim3(NTOK / 8), dim3(256), 0, stream>>>(
        x, routerW + (long)l * DV * EV, e1, e2, g1, g2, acc);
    k_scan<<<1, 32, 0, stream>>>(e1, e2, p1, p2);
    k_zero<<<dim3(1024), dim3(256), 0, stream>>>(xe, (long)BV * EV * CV * DV);
    k_gather<<<dim3(NTOK * 2), dim3(256), 0, stream>>>(x, e1, e2, p1, p2, xe);
    // expert qkv (batched over B*E=32, weights indexed by z%E; M=320 ragged)
    k_gemm<<<dim3(D3 / 128, (CV + 127) / 128, BV * EV), dim3(256), 0, stream>>>(
        xe, eWqkv + (long)l * EV * D3 * DV, ebqkv + (long)l * EV * D3, nullptr,
        qkv, CV, D3, DV, (long)CV * DV, (long)D3 * DV, (long)D3, (long)CV * D3,
        EV);
    k_attn<<<dim3(CV / 16, HV, BV * EV), dim3(32), 0, stream>>>(
        qkv, att, CV, (long)CV * D3, (long)CV * DV, 0);
    k_gemm<<<dim3(DV / 128, (CV + 127) / 128, BV * EV), dim3(256), 0, stream>>>(
        att, eWo + (long)l * EV * DV * DV, ebo + (long)l * EV * DV, nullptr, ey,
        CV, DV, DV, (long)CV * DV, (long)DV * DV, (long)DV, (long)CV * DV, EV);
    // gated scatter-combine + residual ; x = LN2(t)
    k_combine<<<dim3(NTOK), dim3(256), 0, stream>>>(x, ey, e1, e2, p1, p2, g1,
                                                    g2, t);
    k_ln<<<dim3(NTOK / 8), dim3(256), 0, stream>>>(t, n2g + (long)l * DV,
                                                   n2b + (long)l * DV, x);
    k_aux<<<1, 1, 0, stream>>>(acc, auxp);
  }
  // vocab head: logits = x @ headW^T + headb  (the 201 GFLOP / 524 MB one)
  k_gemm<<<dim3(VV / 128, NTOK / 128, 1), dim3(256), 0, stream>>>(
      x, headW, headb, nullptr, out, NTOK, VV, DV, 0, 0, 0, 0, 1);
  k_auxstore<<<1, 1, 0, stream>>>(auxp, out + (long)NTOK * VV);
}
